// NERRelationModel_24257975287949
// MI455X (gfx1250) — matrix operand order repaired and checked by reading, not compiled, not run on hardware
//
#include <hip/hip_runtime.h>
#include <hip/hip_bf16.h>

typedef __attribute__((ext_vector_type(16))) _Float16 v16h;
typedef __attribute__((ext_vector_type(8)))  float    v8f;
typedef __attribute__((ext_vector_type(4)))  float    f32x4;
typedef __attribute__((ext_vector_type(4)))  int      i32x4;

#define BB    32
#define SS    512
#define HH    768
#define NN    4096
#define EE    131072
#define PP    65536
#define TE    (EE + NN)   // edges + self loops
#define H1    4           // heads layer1
#define C1q   128
#define HC1   (H1*C1q)    // 512
#define H2h   1
#define C2q   64
#define HC2   (H2h*C2q)   // 64
#define RHID  256
#define NERH  256
#define NLAB  9

// ---------------- CDNA5 async global->LDS copy (probed) --------------------
#if defined(__has_builtin)
#  if __has_builtin(__builtin_amdgcn_global_load_async_to_lds_b128)
#    define HAVE_ASYNC_LDS 1
#  endif
#endif
#ifndef HAVE_ASYNC_LDS
#  define HAVE_ASYNC_LDS 0
#endif

__device__ __forceinline__ void async_cp16(const float* g, float* l) {
#if HAVE_ASYNC_LDS
  // signature (probe-verified): (global int4*, shared int4*, int offset, int cpol)
  __builtin_amdgcn_global_load_async_to_lds_b128(
      (__attribute__((address_space(1))) i32x4*)((i32x4*)(void*)g),
      (__attribute__((address_space(3))) i32x4*)((i32x4*)(void*)l), 0, 0);
#else
  *(f32x4*)l = *(const f32x4*)g;
#endif
}
__device__ __forceinline__ void async_wait() {
#if HAVE_ASYNC_LDS
  asm volatile("s_wait_asynccnt 0x0" ::: "memory");
#endif
}

// ---------------- monotonic float <-> uint encoding for atomicMax ----------
__device__ __forceinline__ unsigned enc_f(float x) {
  unsigned u = __float_as_uint(x);
  return (u & 0x80000000u) ? ~u : (u | 0x80000000u);
}
__device__ __forceinline__ float dec_f(unsigned e) {
  return __uint_as_float((e & 0x80000000u) ? (e & 0x7fffffffu) : ~e);
}

// ---------------- generic WMMA GEMM: C = act(A[MxK] @ B[KxN] (+bias)) ------
// block = 128 threads (4 waves), block tile = 32 rows x 64 cols.
// Each wave: two stacked 16x16 output tiles (shared B fragment).
// K stepped by 64: async-stage A(32x64) + B(64x64) f32 tiles in LDS,
// 4x v_wmma_f32_16x16x32_f16 per iteration. M%32==0, K%64==0, N%64==0.
__global__ __launch_bounds__(128)
void gemm_wmma(const float* __restrict__ A, const float* __restrict__ B,
               const float* __restrict__ bias, float* __restrict__ C,
               int M, int N, int K, int relu) {
  __shared__ float sA[32][68];     // padded rows: conflict-free, 16B-aligned
  __shared__ float sB[64][68];
  const int tid  = threadIdx.x;
  const int wave = tid >> 5;
  const int lane = tid & 31;
  const int m0    = blockIdx.y * 32;
  const int n0blk = blockIdx.x * 64;
  const int row = lane & 15;
  const int kb  = (lane >> 4) << 3;   // A fragment K sub-base (0 or 8)
  const int ks  = (lane >> 4) << 4;   // unused; B fragment also uses kb (0 or 8)
  const int cl  = (wave << 4) + (lane & 15);
  v8f acc0 = {}, acc1 = {};
  for (int k0 = 0; k0 < K; k0 += 64) {
    // stage A: 32x64 f32 = 512 x 16B chunks
#pragma unroll
    for (int j = 0; j < 4; ++j) {
      int idx = j * 128 + tid;
      int r = idx >> 4, c4 = idx & 15;
      async_cp16(&A[(long)(m0 + r) * K + k0 + c4 * 4], &sA[r][c4 * 4]);
    }
    // stage B: 64x64 f32 = 1024 x 16B chunks
#pragma unroll
    for (int j = 0; j < 8; ++j) {
      int idx = j * 128 + tid;
      int r = idx >> 4, c4 = idx & 15;
      async_cp16(&B[(long)(k0 + r) * N + n0blk + c4 * 4], &sB[r][c4 * 4]);
    }
    async_wait();
    __syncthreads();
#pragma unroll
    for (int s = 0; s < 2; ++s) {
      const int kk = s << 5;
      v16h a0, a1, b;
#pragma unroll
      for (int j = 0; j < 8; ++j) {
        a0[j]     = (_Float16)sA[row][kk + kb + j];
        a0[j + 8] = (_Float16)sA[row][kk + 16 + kb + j];
        a1[j]     = (_Float16)sA[16 + row][kk + kb + j];
        a1[j + 8] = (_Float16)sA[16 + row][kk + 16 + kb + j];
      }
#pragma unroll
      for (int j = 0; j < 16; ++j)
        b[j] = (_Float16)sB[kk + (j & 8) * 2 + kb + (j & 7)][cl];
      acc0 = __builtin_amdgcn_wmma_f32_16x16x32_f16(false, a0, false, b,
                                                    (short)0, acc0, false, false);
      acc1 = __builtin_amdgcn_wmma_f32_16x16x32_f16(false, a1, false, b,
                                                    (short)0, acc1, false, false);
    }
    __syncthreads();
  }
  const int col   = n0blk + cl;
  const int rbase = (lane >> 4) << 3;
  const float bv = bias ? bias[col] : 0.0f;
#pragma unroll
  for (int i = 0; i < 8; ++i) {
    float v0 = acc0[i] + bv;
    float v1 = acc1[i] + bv;
    if (relu) { v0 = fmaxf(v0, 0.0f); v1 = fmaxf(v1, 0.0f); }
    C[(long)(m0 + rbase + i) * N + col]      = v0;
    C[(long)(m0 + 16 + rbase + i) * N + col] = v1;
  }
  (void)M;
}

// ---------------- entity span mean pooling ---------------------------------
__global__ __launch_bounds__(256)
void span_pool(const float* __restrict__ seq, const int* __restrict__ start,
               const int* __restrict__ len, const int* __restrict__ batch,
               float* __restrict__ x) {
  const int n = blockIdx.x;
  const int s0 = start[n], L = len[n], b = batch[n];
  const float* base = seq + ((long)b * SS + s0) * HH;
  const float inv = 1.0f / (float)(L + 1);
  for (int d = threadIdx.x; d < HH; d += 256) {
    float acc = 0.0f;
    for (int s = 0; s <= L; ++s) acc += base[(long)s * HH + d];
    x[(long)n * HH + d] = acc * inv;
  }
}

// ---------------- per-node attention logits --------------------------------
__global__ __launch_bounds__(256)
void gat_node_logits(const float* __restrict__ h, const float* __restrict__ a_src,
                     const float* __restrict__ a_dst, float* __restrict__ es,
                     float* __restrict__ ed, int heads, int C) {
  int t = blockIdx.x * 256 + threadIdx.x;
  if (t >= NN * heads) return;
  int n = t / heads, hd = t % heads;
  const float* hp = h + (long)n * heads * C + (long)hd * C;
  float s = 0.0f, d = 0.0f;
  for (int c = 0; c < C; ++c) {
    float v = hp[c];
    s += v * a_src[hd * C + c];
    d += v * a_dst[hd * C + c];
  }
  es[t] = s;
  ed[t] = d;
}

// ---------------- zero fill -------------------------------------------------
__global__ void zero_f32(float* __restrict__ p, long n) {
  long i = (long)blockIdx.x * 256 + threadIdx.x;
  if (i < n) p[i] = 0.0f;
}

// ---------------- edge passes ----------------------------------------------
__device__ __forceinline__ void edge_sd(const int* ei, int e, int& src, int& dst) {
  if (e < EE) { src = ei[e]; dst = ei[EE + e]; }
  else        { src = dst = e - EE; }
}

__global__ __launch_bounds__(256)
void gat_edge_max(const int* __restrict__ ei, const float* __restrict__ es,
                  const float* __restrict__ ed, unsigned* __restrict__ m, int heads) {
  int t = blockIdx.x * 256 + threadIdx.x;
  if (t >= TE * heads) return;
  int e = t / heads, hd = t % heads;
  int src, dst; edge_sd(ei, e, src, dst);
  float v = es[src * heads + hd] + ed[dst * heads + hd];
  v = v > 0.0f ? v : 0.2f * v;                       // leaky relu
  atomicMax(&m[dst * heads + hd], enc_f(v));
}

__global__ __launch_bounds__(256)
void gat_edge_exp(const int* __restrict__ ei, const float* __restrict__ es,
                  const float* __restrict__ ed, const unsigned* __restrict__ m,
                  float* __restrict__ alpha, float* __restrict__ denom, int heads) {
  int t = blockIdx.x * 256 + threadIdx.x;
  if (t >= TE * heads) return;
  int e = t / heads, hd = t % heads;
  int src, dst; edge_sd(ei, e, src, dst);
  float v = es[src * heads + hd] + ed[dst * heads + hd];
  v = v > 0.0f ? v : 0.2f * v;
  float ex = __expf(v - dec_f(m[dst * heads + hd]));
  alpha[t] = ex;
  atomicAdd(&denom[dst * heads + hd], ex);
}

// one block per edge, heads*C threads
__global__ void gat_edge_agg(const int* __restrict__ ei, const float* __restrict__ h,
                             const float* __restrict__ alpha, const float* __restrict__ denom,
                             float* __restrict__ agg, int heads, int C) {
  int e = blockIdx.x;
  int c = threadIdx.x;                 // 0 .. heads*C-1
  int src, dst; edge_sd(ei, e, src, dst);
  int hd = c / C;
  float w = alpha[(long)e * heads + hd] / denom[dst * heads + hd];
  atomicAdd(&agg[(long)dst * heads * C + c], w * h[(long)src * heads * C + c]);
}

// ---------------- bias (+relu), in place -----------------------------------
__global__ __launch_bounds__(256)
void bias_act(float* __restrict__ p, const float* __restrict__ b, long n, int hc, int relu) {
  long i = (long)blockIdx.x * 256 + threadIdx.x;
  if (i >= n) return;
  float v = p[i] + b[i % hc];
  if (relu) v = fmaxf(v, 0.0f);
  p[i] = v;
}

// ---------------- fused relation head --------------------------------------
// grid (P/16, 6); 512 threads = 16 waves. Per block: async-gather 16 pair rows,
// async-stage relW1 K-slab, WMMA 16x256 hidden, LayerNorm+ReLU+256->1 proj.
__global__ __launch_bounds__(512)
void relation_head(const float* __restrict__ x2, const int* __restrict__ pair_idx,
                   const float* __restrict__ relW1, const float* __restrict__ relb1,
                   const float* __restrict__ ln_g, const float* __restrict__ ln_b,
                   const float* __restrict__ relW2, const float* __restrict__ relb2,
                   float* __restrict__ out) {
  __shared__ float sPF[16][132];       // 16 x 128 pair features (padded)
  __shared__ float sBr[32][260];       // 32 x 256 weight slab (padded)
  __shared__ float sHid[16][RHID];     // 16 x 256 hidden
  const int tid  = threadIdx.x;
  const int wave = tid >> 5;
  const int lane = tid & 31;
  const int p0 = blockIdx.x * 16;
  const int r  = blockIdx.y;
  // gather pair features: cat(x2[e1], x2[e2]); one 16B async gather per thread
  {
    int rowp = tid >> 5, c4 = tid & 31;
    int pa = pair_idx[(p0 + rowp) * 2 + (c4 >> 4)];
    async_cp16(&x2[(long)pa * HC2 + (c4 & 15) * 4], &sPF[rowp][c4 * 4]);
  }
  const float* W1r = relW1 + (long)r * 128 * RHID;
  const int row  = lane & 15;
  const int kb   = (lane >> 4) << 3;
  const int ks   = (lane >> 4) << 4;
  const int coll = (wave << 4) + (lane & 15);
  v8f acc = {};
  for (int k0 = 0; k0 < 128; k0 += 32) {
    // stage B slab: 32x256 f32 = 2048 x 16B chunks, 512 threads x 4
#pragma unroll
    for (int j = 0; j < 4; ++j) {
      int idx = j * 512 + tid;
      int rr = idx >> 6, c4 = idx & 63;
      async_cp16(&W1r[(long)(k0 + rr) * RHID + c4 * 4], &sBr[rr][c4 * 4]);
    }
    async_wait();
    __syncthreads();
    v16h a, b;
#pragma unroll
    for (int j = 0; j < 8; ++j) {
      a[j]     = (_Float16)sPF[row][k0 + kb + j];
      a[j + 8] = (_Float16)sPF[row][k0 + 16 + kb + j];
    }
#pragma unroll
    for (int j = 0; j < 16; ++j)
      b[j] = (_Float16)sBr[(j & 8) * 2 + kb + (j & 7)][coll];
    acc = __builtin_amdgcn_wmma_f32_16x16x32_f16(false, a, false, b,
                                                 (short)0, acc, false, false);
    __syncthreads();
  }
  const int rbase = (lane >> 4) << 3;
#pragma unroll
  for (int i = 0; i < 8; ++i)
    sHid[rbase + i][coll] = acc[i] + relb1[r * RHID + coll];
  __syncthreads();
  // wave w: LayerNorm + ReLU + dot(relW2) for pair row w
  float s = 0.0f, ss = 0.0f, vals[8];
#pragma unroll
  for (int j = 0; j < 8; ++j) {
    float v = sHid[wave][lane * 8 + j];
    vals[j] = v; s += v; ss += v * v;
  }
#pragma unroll
  for (int off = 16; off >= 1; off >>= 1) {
    s  += __shfl_xor(s, off, 32);
    ss += __shfl_xor(ss, off, 32);
  }
  float mu  = s * (1.0f / RHID);
  float var = ss * (1.0f / RHID) - mu * mu;
  float rs  = rsqrtf(var + 1e-5f);
  float dot = 0.0f;
#pragma unroll
  for (int j = 0; j < 8; ++j) {
    int c = lane * 8 + j;
    float v = (vals[j] - mu) * rs * ln_g[r * RHID + c] + ln_b[r * RHID + c];
    v = fmaxf(v, 0.0f);
    dot += v * relW2[r * RHID + c];
  }
#pragma unroll
  for (int off = 16; off >= 1; off >>= 1) dot += __shfl_xor(dot, off, 32);
  if (lane == 0) out[(long)r * PP + p0 + wave] = dot + relb2[r];
}

// ---------------- NER second projection [*,256] -> [*,9] -------------------
__global__ __launch_bounds__(256)
void ner_out(const float* __restrict__ hid, const float* __restrict__ W2,
             const float* __restrict__ b2, float* __restrict__ out) {
  int t = blockIdx.x * 256 + threadIdx.x;
  if (t >= BB * SS * NLAB) return;
  int rowi = t / NLAB, l = t % NLAB;
  const float* hp = hid + (long)rowi * NERH;
  float acc = b2[l];
  for (int k = 0; k < NERH; ++k) acc += hp[k] * W2[k * NLAB + l];
  out[t] = acc;
}

// ---------------------------------------------------------------------------
extern "C" void kernel_launch(void* const* d_in, const int* in_sizes, int n_in,
                              void* d_out, int out_size, void* d_ws, size_t ws_size,
                              hipStream_t stream) {
  const float* seq       = (const float*)d_in[0];
  const int*   ent_start = (const int*)d_in[1];
  const int*   ent_len   = (const int*)d_in[2];
  const int*   ent_batch = (const int*)d_in[3];
  const int*   edge_idx  = (const int*)d_in[4];
  const int*   pair_idx  = (const int*)d_in[5];
  const float* W1        = (const float*)d_in[6];
  const float* a_src1    = (const float*)d_in[7];
  const float* a_dst1    = (const float*)d_in[8];
  const float* b1        = (const float*)d_in[9];
  const float* W2        = (const float*)d_in[10];
  const float* a_src2    = (const float*)d_in[11];
  const float* a_dst2    = (const float*)d_in[12];
  const float* b2        = (const float*)d_in[13];
  const float* relW1     = (const float*)d_in[14];
  const float* relb1     = (const float*)d_in[15];
  const float* ln_g      = (const float*)d_in[16];
  const float* ln_b      = (const float*)d_in[17];
  const float* relW2     = (const float*)d_in[18];
  const float* relb2     = (const float*)d_in[19];
  const float* nerW1     = (const float*)d_in[20];
  const float* nerb1     = (const float*)d_in[21];
  const float* nerW2     = (const float*)d_in[22];
  const float* nerb2     = (const float*)d_in[23];

  float* out_ner = (float*)d_out;                  // [32*512*9]
  float* out_rel = (float*)d_out + BB * SS * NLAB; // [6*65536]

  // ---- bump allocator on ws ----
  char* wsb = (char*)d_ws;
  size_t off = 0;
  auto alloc = [&](size_t bytes) -> void* {
    void* p = wsb + off;
    off += (bytes + 255) & ~(size_t)255;
    return p;
  };
  float*    x      = (float*)alloc((size_t)NN * HH * 4);   // entity feats (reused for NER hidden)
  float*    h1     = (float*)alloc((size_t)NN * HC1 * 4);
  float*    es1    = (float*)alloc((size_t)NN * H1 * 4);
  float*    ed1    = (float*)alloc((size_t)NN * H1 * 4);
  unsigned* m1     = (unsigned*)alloc((size_t)NN * H1 * 4);
  float*    den1   = (float*)alloc((size_t)NN * H1 * 4);
  float*    alpha1 = (float*)alloc((size_t)TE * H1 * 4);
  float*    agg1   = (float*)alloc((size_t)NN * HC1 * 4);  // becomes x1 in-place
  float*    h2     = (float*)alloc((size_t)NN * HC2 * 4);
  float*    es2    = (float*)alloc((size_t)NN * H2h * 4);
  float*    ed2    = (float*)alloc((size_t)NN * H2h * 4);
  unsigned* m2     = (unsigned*)alloc((size_t)NN * H2h * 4);
  float*    den2   = (float*)alloc((size_t)NN * H2h * 4);
  float*    alpha2 = (float*)alloc((size_t)TE * H2h * 4);
  float*    agg2   = (float*)alloc((size_t)NN * HC2 * 4);  // becomes x2 in-place
  float*    hidner = x;   // reuse x..h1 region (free after GAT linears) for 16384x256

  // 1) entity span mean pooling
  span_pool<<<NN, 256, 0, stream>>>(seq, ent_start, ent_len, ent_batch, x);

  // 2) GAT1 linear: h1 = x @ W1   [4096,768]x[768,512]
  gemm_wmma<<<dim3(HC1 / 64, NN / 32), 128, 0, stream>>>(x, W1, nullptr, h1, NN, HC1, HH, 0);

  // 3) attention logits layer 1
  gat_node_logits<<<(NN * H1 + 255) / 256, 256, 0, stream>>>(h1, a_src1, a_dst1, es1, ed1, H1, C1q);

  // 4) init m/denom/agg
  zero_f32<<<(NN * H1 + 255) / 256, 256, 0, stream>>>((float*)m1, (long)NN * H1);
  zero_f32<<<(NN * H1 + 255) / 256, 256, 0, stream>>>(den1, (long)NN * H1);
  zero_f32<<<((long)NN * HC1 + 255) / 256, 256, 0, stream>>>(agg1, (long)NN * HC1);

  // 5-7) segment max / exp+denom / weighted aggregation
  gat_edge_max<<<(TE * H1 + 255) / 256, 256, 0, stream>>>(edge_idx, es1, ed1, m1, H1);
  gat_edge_exp<<<(TE * H1 + 255) / 256, 256, 0, stream>>>(edge_idx, es1, ed1, m1, alpha1, den1, H1);
  gat_edge_agg<<<TE, HC1, 0, stream>>>(edge_idx, h1, alpha1, den1, agg1, H1, C1q);

  // 8) x1 = relu(agg1 + b1), in place
  bias_act<<<((long)NN * HC1 + 255) / 256, 256, 0, stream>>>(agg1, b1, (long)NN * HC1, HC1, 1);

  // 9) GAT2 linear: h2 = x1 @ W2   [4096,512]x[512,64]
  gemm_wmma<<<dim3(HC2 / 64, NN / 32), 128, 0, stream>>>(agg1, W2, nullptr, h2, NN, HC2, HC1, 0);

  // 10-15) layer-2 attention + aggregation + bias
  gat_node_logits<<<(NN * H2h + 255) / 256, 256, 0, stream>>>(h2, a_src2, a_dst2, es2, ed2, H2h, C2q);
  zero_f32<<<(NN * H2h + 255) / 256, 256, 0, stream>>>((float*)m2, (long)NN * H2h);
  zero_f32<<<(NN * H2h + 255) / 256, 256, 0, stream>>>(den2, (long)NN * H2h);
  zero_f32<<<((long)NN * HC2 + 255) / 256, 256, 0, stream>>>(agg2, (long)NN * HC2);
  gat_edge_max<<<(TE * H2h + 255) / 256, 256, 0, stream>>>(edge_idx, es2, ed2, m2, H2h);
  gat_edge_exp<<<(TE * H2h + 255) / 256, 256, 0, stream>>>(edge_idx, es2, ed2, m2, alpha2, den2, H2h);
  gat_edge_agg<<<TE, HC2, 0, stream>>>(edge_idx, h2, alpha2, den2, agg2, H2h, C2q);
  bias_act<<<((long)NN * HC2 + 255) / 256, 256, 0, stream>>>(agg2, b2, (long)NN * HC2, HC2, 0);

  // 16) fused relation head -> rel_logits [6,65536]
  relation_head<<<dim3(PP / 16, 6), 512, 0, stream>>>(agg2, pair_idx, relW1, relb1,
                                                      ln_g, ln_b, relW2, relb2, out_rel);

  // 17) NER GEMM1 (+bias+relu): [16384,768]x[768,256]  (reuses freed ws region)
  gemm_wmma<<<dim3(NERH / 64, (BB * SS) / 32), 128, 0, stream>>>(seq, nerW1, nerb1, hidner,
                                                                 BB * SS, NERH, HH, 1);
  // 18) NER projection to 9 labels
  ner_out<<<(BB * SS * NLAB + 255) / 256, 256, 0, stream>>>(hidner, nerW2, nerb2, out_ner);

  (void)in_sizes; (void)n_in; (void)out_size; (void)ws_size;
}
